// QKVAttention_89721866813760
// MI455X (gfx1250) — compile-verified, matrix operand order checked by reading, not run on hardware
//
#include <hip/hip_runtime.h>

// QKV attention (flash-style) for MI455X / gfx1250, wave32 + WMMA.
// qkv: [B, 3*C, T] fp32, C=64, T=2048 -> out: [B, C, T] fp32.
//
// Key trick: compute S^T = K_tile x Q so each lane's column is ONE query and
// its 8 accumulator rows are keys. Then:
//   - softmax max: per-lane max + a single shfl_xor(16)   (no bpermute storm)
//   - P = exp(S^T) is ALREADY in B-fragment layout for O^T = V x P
//     -> P never round-trips through LDS
//   - row sums via ones(A) x P(B) WMMA
// 8 waves x 2 query tiles = 256 queries/WG -> 4x less K/V L2 re-read traffic.

typedef __attribute__((ext_vector_type(4)))  _Float16 v4h;
typedef __attribute__((ext_vector_type(8)))  _Float16 v8h;
typedef __attribute__((ext_vector_type(16))) _Float16 v16h;
typedef __attribute__((ext_vector_type(8)))  float    v8f;

#define T_SEQ 2048
#define CH    64
#define KBLK  64               // keys per block iteration
#define WAVES 8
#define QTPW  2                // query tiles (of 16) per wave
#define NTHR  (WAVES * 32)     // 256
#define QPWG  (WAVES * QTPW * 16)  // 256 queries per workgroup
#define LDSW  (CH + 8)         // 72-half row stride (144B): 16B-aligned rows,
                               // 36-dword lane stride -> conflict-free b128

__device__ __forceinline__ v16h cat8(v8h lo, v8h hi) {
    return __builtin_shufflevector(lo, hi, 0,1,2,3,4,5,6,7,8,9,10,11,12,13,14,15);
}
__device__ __forceinline__ v4h pk4(float a, float b, float c, float d) {
    v4h w; w[0] = (_Float16)a; w[1] = (_Float16)b; w[2] = (_Float16)c; w[3] = (_Float16)d;
    return w;
}
#define WMMA_F16(A, B, C) \
    __builtin_amdgcn_wmma_f32_16x16x32_f16(false, (A), false, (B), (short)0, (C), false, false)

__global__ __launch_bounds__(NTHR)
void qkv_attention_flash(const float* __restrict__ qkv, float* __restrict__ out) {
    __shared__ __align__(16) _Float16 k_sh[KBLK][LDSW];   // [key][channel]
    __shared__ __align__(16) _Float16 v_sh[CH][LDSW];     // [channel][key]

    const int QBLKS = T_SEQ / QPWG;              // 8 query blocks per batch
    const int b    = blockIdx.x / QBLKS;
    const int q0   = (blockIdx.x % QBLKS) * QPWG;
    const int tid  = threadIdx.x;
    const int wv   = tid >> 5;
    const int lane = tid & 31;
    const int col  = lane & 15;                  // fragment M (A) / N (B,C,D)
    const int grp  = lane >> 4;                  // half-wave group
    const int tq   = q0 + wv * (QTPW * 16);      // first query of this wave

    const size_t bbase = (size_t)b * 3 * CH * T_SEQ;
    const float* qp = qkv + bbase;                            // Q [c][t]
    const float* kp = qkv + bbase + (size_t)CH * T_SEQ;       // K [c][s]
    const float* vp = qkv + bbase + (size_t)2 * CH * T_SEQ;   // V [c][s]

    // Fold softmax scale 1/sqrt(64) AND log2(e) into Q once: exps become bare
    // v_exp_f32 (exp2) with no multiplies in the hot loop.
    const float qscale = 0.125f * 1.44269504088896340736f;

    // ---- Q B-fragments (col N = query, per-lane K = channel pattern
    // K = (e&7) + 8*grp + 16*(e>=8)), resident in VGPRs.
    v16h bq[QTPW][2];
#pragma unroll
    for (int qt = 0; qt < QTPW; ++qt)
#pragma unroll
        for (int e = 0; e < 16; ++e) {
            const int k = (e & 7) + (grp << 3) + ((e >> 3) << 4);
            const int t = tq + qt * 16 + col;
            bq[qt][0][e] = (_Float16)(qp[(size_t)k        * T_SEQ + t] * qscale);
            bq[qt][1][e] = (_Float16)(qp[(size_t)(k + 32) * T_SEQ + t] * qscale);
        }
    v16h ones;
#pragma unroll
    for (int e = 0; e < 16; ++e) ones[e] = (_Float16)1.0f;

    // ---- flash state. O is accumulated TRANSPOSED: D[ch][query]; each lane's
    // column (n = col) is its query, rows r -> channel (r + 8*grp) per tile.
    v8f zero = {};
    v8f acc_o[QTPW][4] = {{zero, zero, zero, zero}, {zero, zero, zero, zero}};
    v8f acc_l[QTPW]    = {zero, zero};           // per-query sums (all rows equal)
    float q_m[QTPW]    = {-1e30f, -1e30f};       // per-lane running max (scalar!)

    for (int s0 = 0; s0 < T_SEQ; s0 += KBLK) {
        // ---- stage K transposed -> k_sh[key][ch] f16 (4x4 micro-tiles)
        {
            const int s  = (tid & 15) * 4;               // 256 tiles, 1/thread
            const int c0 = (tid >> 4) * 4;
            const float4 r0 = *(const float4*)(kp + (size_t)(c0 + 0) * T_SEQ + s0 + s);
            const float4 r1 = *(const float4*)(kp + (size_t)(c0 + 1) * T_SEQ + s0 + s);
            const float4 r2 = *(const float4*)(kp + (size_t)(c0 + 2) * T_SEQ + s0 + s);
            const float4 r3 = *(const float4*)(kp + (size_t)(c0 + 3) * T_SEQ + s0 + s);
            *(v4h*)&k_sh[s + 0][c0] = pk4(r0.x, r1.x, r2.x, r3.x);
            *(v4h*)&k_sh[s + 1][c0] = pk4(r0.y, r1.y, r2.y, r3.y);
            *(v4h*)&k_sh[s + 2][c0] = pk4(r0.z, r1.z, r2.z, r3.z);
            *(v4h*)&k_sh[s + 3][c0] = pk4(r0.w, r1.w, r2.w, r3.w);
        }
        // ---- stage V -> v_sh[ch][key] f16 (no transpose)
#pragma unroll
        for (int it = 0; it < 4; ++it) {
            const int chunk = tid + it * NTHR;           // 1024 float4 chunks
            const int ch = chunk >> 4;
            const int sg = (chunk & 15) * 4;
            const float4 r = *(const float4*)(vp + (size_t)ch * T_SEQ + s0 + sg);
            *(v4h*)&v_sh[ch][sg] = pk4(r.x, r.y, r.z, r.w);
        }
        if (s0 + KBLK < T_SEQ) {                 // global_prefetch_b8 next block
            const int ch = tid >> 2, s = (tid & 3) * 16;
            __builtin_prefetch(kp + (size_t)ch * T_SEQ + s0 + KBLK + s, 0, 3);
            __builtin_prefetch(vp + (size_t)ch * T_SEQ + s0 + KBLK + s, 0, 3);
        }
        __syncthreads();

        // ---- S^T = K_tile(A) x Q(B): s_acc[qt][kt] holds keys kt*16+r+8*grp
        // for query tq+qt*16+col. K A-fragments shared by both query tiles.
        v8f s_acc[QTPW][4];
#pragma unroll
        for (int kt = 0; kt < 4; ++kt) {
            const _Float16* kr = &k_sh[kt * 16 + col][0];
            const v16h ak0 = cat8(*(const v8h*)(kr + 8 * grp),
                                  *(const v8h*)(kr + 16 + 8 * grp));
            const v16h ak1 = cat8(*(const v8h*)(kr + 32 + 8 * grp),
                                  *(const v8h*)(kr + 48 + 8 * grp));
#pragma unroll
            for (int qt = 0; qt < QTPW; ++qt) {
                v8f c = {};
                c = WMMA_F16(ak0, bq[qt][0], c);
                c = WMMA_F16(ak1, bq[qt][1], c);
                s_acc[qt][kt] = c;
            }
        }

        // ---- online softmax + build P B-fragments entirely in registers
        v16h bpf[QTPW][2];
#pragma unroll
        for (int qt = 0; qt < QTPW; ++qt) {
            float mx = s_acc[qt][0][0];
#pragma unroll
            for (int kt = 0; kt < 4; ++kt)
#pragma unroll
                for (int r = 0; r < 8; ++r)
                    mx = fmaxf(mx, s_acc[qt][kt][r]);
            mx = fmaxf(mx, __shfl_xor(mx, 16, 32));  // partner holds other 8 keys
            const float m_new = fmaxf(q_m[qt], mx);
            const float alpha = __builtin_amdgcn_exp2f(q_m[qt] - m_new);
            q_m[qt] = m_new;
#pragma unroll
            for (int r = 0; r < 8; ++r) acc_l[qt][r] *= alpha;
#pragma unroll
            for (int ct = 0; ct < 4; ++ct)
#pragma unroll
                for (int r = 0; r < 8; ++r) acc_o[qt][ct][r] *= alpha;

            // B-fragment of P: element e<8 <-> tile kt_lo key (e+8*grp),
            // e>=8 <-> tile kt_hi -- exactly this lane's s_acc values.
            v16h bp01, bp23;
#pragma unroll
            for (int e = 0; e < 8; ++e) {
                bp01[e]     = (_Float16)__builtin_amdgcn_exp2f(s_acc[qt][0][e] - m_new);
                bp01[e + 8] = (_Float16)__builtin_amdgcn_exp2f(s_acc[qt][1][e] - m_new);
                bp23[e]     = (_Float16)__builtin_amdgcn_exp2f(s_acc[qt][2][e] - m_new);
                bp23[e + 8] = (_Float16)__builtin_amdgcn_exp2f(s_acc[qt][3][e] - m_new);
            }
            // per-query sums: acc_l += ones(A) x P(B)
            acc_l[qt] = WMMA_F16(ones, bp01, acc_l[qt]);
            acc_l[qt] = WMMA_F16(ones, bp23, acc_l[qt]);
            bpf[qt][0] = bp01;
            bpf[qt][1] = bp23;
        }

        // ---- O^T += V(A) x P(B): V A-fragments shared by both query tiles
#pragma unroll
        for (int ct = 0; ct < 4; ++ct) {
            const _Float16* vr = &v_sh[ct * 16 + col][0];
            const v16h av0 = cat8(*(const v8h*)(vr + 8 * grp),
                                  *(const v8h*)(vr + 16 + 8 * grp));
            const v16h av1 = cat8(*(const v8h*)(vr + 32 + 8 * grp),
                                  *(const v8h*)(vr + 48 + 8 * grp));
#pragma unroll
            for (int qt = 0; qt < QTPW; ++qt) {
                acc_o[qt][ct] = WMMA_F16(av0, bpf[qt][0], acc_o[qt][ct]);
                acc_o[qt][ct] = WMMA_F16(av1, bpf[qt][1], acc_o[qt][ct]);
            }
        }
        __syncthreads();   // protect k_sh/v_sh for next block
    }

    // ---- epilogue: lane's column is query t = tq + qt*16 + col;
    // rows are channels ct*16 + r + 8*grp. Half-wave stores are 64B runs.
    float* op = out + (size_t)b * CH * T_SEQ;
#pragma unroll
    for (int qt = 0; qt < QTPW; ++qt) {
        const float inv = 1.0f / acc_l[qt][0];
        const int t = tq + qt * 16 + col;
#pragma unroll
        for (int ct = 0; ct < 4; ++ct)
#pragma unroll
            for (int r = 0; r < 8; ++r) {
                const int c = ct * 16 + r + (grp << 3);
                op[(size_t)c * T_SEQ + t] = acc_o[qt][ct][r] * inv;
            }
    }
}

extern "C" void kernel_launch(void* const* d_in, const int* in_sizes, int n_in,
                              void* d_out, int out_size, void* d_ws, size_t ws_size,
                              hipStream_t stream) {
    const float* qkv = (const float*)d_in[0];
    float* out = (float*)d_out;
    const int B = in_sizes[0] / (3 * CH * T_SEQ);          // 64
    dim3 grid(B * (T_SEQ / QPWG));                         // 512 workgroups
    qkv_attention_flash<<<grid, NTHR, 0, stream>>>(qkv, out);
}
